// MultiHeadAttention_21973052686482
// MI455X (gfx1250) — matrix operand order repaired and checked by reading, not compiled, not run on hardware
//
#include <hip/hip_runtime.h>
#include <hip/hip_bf16.h>

// ---------------------------------------------------------------------------
// MHA forward for MI455X (gfx1250, wave32, WMMA + TDM).
// B=8, S=1024, D=1024, H=16, HD=64.
// d_out = [out: B*S*D f32][attn: B*H*S*S f32]
// ---------------------------------------------------------------------------

typedef __attribute__((ext_vector_type(16))) __bf16 v16bf;
typedef __attribute__((ext_vector_type(8)))  float  v8f;
typedef __attribute__((ext_vector_type(4)))  unsigned int u32x4;
typedef __attribute__((ext_vector_type(2)))  unsigned int u32x2;
typedef __attribute__((ext_vector_type(8)))  int iv8;
typedef __attribute__((ext_vector_type(4)))  int iv4;
typedef __attribute__((ext_vector_type(4)))  unsigned int uv4;
typedef unsigned short u16;

union Frag {
    v16bf bf;
    u32x4 u4[2];
};

__device__ __forceinline__ v8f zero8() {
    v8f z = {0.f, 0.f, 0.f, 0.f, 0.f, 0.f, 0.f, 0.f};
    return z;
}

// f32 -> bf16, round-to-nearest-even
__device__ __forceinline__ u16 f2bf(float x) {
    unsigned int u = __builtin_bit_cast(unsigned int, x);
    unsigned int r = u + 0x7FFFu + ((u >> 16) & 1u);
    return (u16)(r >> 16);
}

__device__ __forceinline__ v8f wmma_bf16(const Frag& a, const Frag& b, v8f c) {
    // (neg_a, A, neg_b, B, c_mod, C, reuse_a, reuse_b)
    return __builtin_amdgcn_wmma_f32_16x16x32_bf16(false, a.bf, false, b.bf,
                                                   (short)0, c, false, false);
}

// ---------------------------------------------------------------------------
// Tensor Data Mover: 2D bf16 tile (tile_w elems/row, tile_h rows, row stride
// in elements) from global -> LDS (rows packed contiguously).
// D# packing per CDNA5 ISA sect 8.3/8.4.
// ---------------------------------------------------------------------------
#if __has_builtin(__builtin_amdgcn_tensor_load_to_lds)
#define HAVE_TDM 1
#else
#define HAVE_TDM 0
#endif

#if HAVE_TDM
__device__ __forceinline__ unsigned lds_addr32(void* p) {
    // generic -> LDS(AS3) pointer -> 32-bit LDS byte offset
    return (unsigned)(unsigned long long)(__attribute__((address_space(3))) char*)p;
}

__device__ __forceinline__ void tdm_load_2d_bf16(unsigned lds_off, const void* gptr,
                                                 unsigned tile_w, unsigned tile_h,
                                                 unsigned row_stride_elems) {
    unsigned long long ga = (unsigned long long)gptr;
    uv4 g0;
    g0[0] = 1u;                                              // count=1, user mode
    g0[1] = lds_off;                                         // lds_addr [63:32]
    g0[2] = (unsigned)(ga & 0xFFFFFFFFu);                    // global_addr lo
    g0[3] = (unsigned)((ga >> 32) & 0x01FFFFFFu) | (2u << 30); // addr hi | type=2

    const unsigned td0 = 1u << 16;   // tensor_dim0 (OOB bound, generous)
    const unsigned td1 = 1u << 20;   // tensor_dim1
    iv8 g1;
    g1[0] = (int)(1u << 16);                                       // data_size=1 (2B), mask=0
    g1[1] = (int)((td0 & 0xFFFFu) << 16);                          // [63:48]=td0.lo16
    g1[2] = (int)(((td0 >> 16) & 0xFFFFu) | ((td1 & 0xFFFFu) << 16));
    g1[3] = (int)(((td1 >> 16) & 0xFFFFu) | ((tile_w & 0xFFFFu) << 16)); // tile_dim0
    g1[4] = (int)(tile_h & 0xFFFFu);                               // tile_dim1, tile_dim2=0
    g1[5] = (int)row_stride_elems;                                 // dim0_stride lo32
    g1[6] = 0;                                                     // stride hi / dim1_stride
    g1[7] = 0;

    iv4 z4 = {0, 0, 0, 0};
#if defined(__clang_major__) && (__clang_major__ >= 23)
    iv8 z8 = {0, 0, 0, 0, 0, 0, 0, 0};
    __builtin_amdgcn_tensor_load_to_lds(g0, g1, z4, z4, z8, 0);
#else
    __builtin_amdgcn_tensor_load_to_lds(g0, g1, z4, z4, 0);
#endif
}
#endif // HAVE_TDM

// ---------------------------------------------------------------------------
// fp32 -> bf16 conversion, 4 elements/thread (b128 in, b64 out)
// ---------------------------------------------------------------------------
__global__ void cvt_f32_bf16(const float* __restrict__ x, u16* __restrict__ y, int n) {
    int i = (blockIdx.x * blockDim.x + threadIdx.x) * 4;
    if (i < n) {
        u32x4 f = *(const u32x4*)(x + i);
        unsigned r0 = f2bf(__builtin_bit_cast(float, f[0]));
        unsigned r1 = f2bf(__builtin_bit_cast(float, f[1]));
        unsigned r2 = f2bf(__builtin_bit_cast(float, f[2]));
        unsigned r3 = f2bf(__builtin_bit_cast(float, f[3]));
        u32x2 o;
        o[0] = r0 | (r1 << 16);
        o[1] = r2 | (r3 << 16);
        *(u32x2*)(y + i) = o;
    }
}

// ---------------------------------------------------------------------------
// Vt[((b*H+h)*64 + d)*S + s] = Vp[(b*S + s)*D + h*64 + d]
// ---------------------------------------------------------------------------
__global__ void transpose_v(const u16* __restrict__ Vp, u16* __restrict__ Vt) {
    int idx = blockIdx.x * blockDim.x + threadIdx.x;   // over 8*16*64*1024
    int s  = idx & 1023;
    int d  = (idx >> 10) & 63;
    int hh = (idx >> 16) & 15;
    int bb = idx >> 20;
    Vt[idx] = Vp[((size_t)(bb * 1024 + s)) * 1024 + hh * 64 + d];
}

// ---------------------------------------------------------------------------
// C = A @ W^T + bias.  A:[M,K] bf16 row-major, W:[N,K] bf16 row-major.
// Block: 256 threads = 8 waves; block tile 256(M) x 64(N); wave tile 32x64.
// W tile (64x32 bf16 = 4KB) staged in LDS by the Tensor Data Mover (wave 0),
// double-buffered, one barrier per K-step:
//   [wait tensorcnt][barrier][issue tile i+1][compute tile i]
// All 4 B-fragments are loaded into distinct registers before the 8 WMMAs so
// the XDL pipe sees 8 independent chains with a single dscnt wait per step.
// ---------------------------------------------------------------------------
__global__ void gemm_nt(const u16* __restrict__ A, const u16* __restrict__ W,
                        const float* __restrict__ bias,
                        u16* __restrict__ Obf, float* __restrict__ Of32,
                        int M, int N, int K) {
    __shared__ u16 wtile[2][64 * 32];   // 8 KB

    const int lane    = threadIdx.x & 31;
    const int w       = threadIdx.x >> 5;
    const int laneRow = lane & 15;
    const int hi      = lane >> 4;

    const int m0 = blockIdx.x * 256 + w * 32;
    const int n0 = blockIdx.y * 64;

    v8f acc[2][4];
#pragma unroll
    for (int i = 0; i < 2; i++)
#pragma unroll
        for (int j = 0; j < 4; j++) acc[i][j] = zero8();

    const u16* pa0_row = A + (size_t)(m0 + laneRow) * K;
    const u16* pa1_row = pa0_row + (size_t)16 * K;
    const u16* wbase   = W + (size_t)n0 * K;

    const int nsteps = K / 32;

#if HAVE_TDM
    if (w == 0)
        tdm_load_2d_bf16(lds_addr32(&wtile[0][0]), wbase, 32, 64, (unsigned)K);
#endif

    for (int i = 0; i < nsteps; ++i) {
        const int kk = i * 32;
#if HAVE_TDM
        if (w == 0) __builtin_amdgcn_s_wait_tensorcnt(0);
        __syncthreads();
        if (w == 0 && i + 1 < nsteps)
            tdm_load_2d_bf16(lds_addr32(&wtile[(i + 1) & 1][0]),
                             wbase + kk + 32, 32, 64, (unsigned)K);
#endif
        // A fragments (K-contiguous per documented 16-bit A layout)
        Frag a0, a1;
        const u16* pa0 = pa0_row + kk + hi * 8;
        const u16* pa1 = pa1_row + kk + hi * 8;
        a0.u4[0] = *(const u32x4*)(pa0);
        a0.u4[1] = *(const u32x4*)(pa0 + 16);
        a1.u4[0] = *(const u32x4*)(pa1);
        a1.u4[1] = *(const u32x4*)(pa1 + 16);
        if (kk + 64 < K) {
            __builtin_prefetch(pa0 + 64, 0, 1);   // -> global_prefetch_b8
            __builtin_prefetch(pa1 + 64, 0, 1);
        }

        // Load all 4 B fragments into distinct registers, then multiply.
        Frag bfr[4];
#if HAVE_TDM
        const u16* wt = &wtile[i & 1][0];
#pragma unroll
        for (int j = 0; j < 4; j++) {
            const u16* pb = wt + (j * 16 + laneRow) * 32 + hi * 8;  // ds_load_b128 x2
            bfr[j].u4[0] = *(const u32x4*)(pb);
            bfr[j].u4[1] = *(const u32x4*)(pb + 16);
        }
#else
#pragma unroll
        for (int j = 0; j < 4; j++) {
            const u16* pb = wbase + (size_t)(j * 16 + laneRow) * K + kk + hi * 8;
            bfr[j].u4[0] = *(const u32x4*)(pb);
            bfr[j].u4[1] = *(const u32x4*)(pb + 16);
        }
#endif
#pragma unroll
        for (int j = 0; j < 4; j++) {
            acc[0][j] = wmma_bf16(a0, bfr[j], acc[0][j]);
            acc[1][j] = wmma_bf16(a1, bfr[j], acc[1][j]);
        }
    }

#pragma unroll
    for (int j = 0; j < 4; j++) {
        const int col = n0 + j * 16 + laneRow;
        const float bv = bias[col];
#pragma unroll
        for (int i = 0; i < 2; i++) {
#pragma unroll
            for (int r = 0; r < 8; r++) {
                // D layout: row = r + 8*hi within 16x16 tile, col = lane&15
                const int row = m0 + i * 16 + r + hi * 8;
                const float v = acc[i][j][r] + bv;
                if (Obf) Obf[(size_t)row * N + col] = f2bf(v);
                else     Of32[(size_t)row * N + col] = v;
            }
        }
    }
}

// ---------------------------------------------------------------------------
// Attention core: one block per (b, h, 32 query rows). 256 threads = 8 waves.
// LDS: 32x1024 f32 scores (128KB) + 32x1024 bf16 probs (64KB) = 192KB.
// Phase 1: S = Q K^T (WMMA bf16) -> LDS
// Phase 2: softmax rows (scale 1/8, mask) -> attn f32 to global, bf16 to LDS
// Phase 3: X = P V (WMMA bf16, 2 accumulator chains for XDL ILP) -> Xc bf16
// ---------------------------------------------------------------------------
__global__ void attn_kernel(const u16* __restrict__ Qp, const u16* __restrict__ Kp,
                            const u16* __restrict__ Vt, const int* __restrict__ mask,
                            float* __restrict__ attn_out, u16* __restrict__ Xc) {
    extern __shared__ char smem_raw[];
    float* sc = (float*)smem_raw;                               // [32][1024] f32
    u16*   pr = (u16*)(smem_raw + 32 * 1024 * sizeof(float));   // [32][1024] bf16

    const int lane    = threadIdx.x & 31;
    const int w       = threadIdx.x >> 5;
    const int laneRow = lane & 15;
    const int hi      = lane >> 4;

    const int bid  = blockIdx.x;          // B*H*(S/32) = 4096 blocks
    const int qblk = bid & 31;
    const int h    = (bid >> 5) & 15;
    const int b    = bid >> 9;
    const int q0   = qblk * 32;

    // ---- Phase 1: scores = Q @ K^T ------------------------------------
    {
        const int qsub = w & 1;
        Frag aq[2];               // d-steps 0 and 32 of head_dim 64
        const u16* pq = Qp + (size_t)(b * 1024 + q0 + qsub * 16 + laneRow) * 1024
                           + h * 64 + hi * 8;
        aq[0].u4[0] = *(const u32x4*)(pq);
        aq[0].u4[1] = *(const u32x4*)(pq + 16);
        aq[1].u4[0] = *(const u32x4*)(pq + 32);
        aq[1].u4[1] = *(const u32x4*)(pq + 48);

        for (int kb = (w >> 1); kb < 64; kb += 4) {   // 16-wide key blocks
            // load both K d-step fragments before the WMMA chain
            Frag bk0, bk1;
            const u16* pk = Kp + (size_t)(b * 1024 + kb * 16 + laneRow) * 1024
                               + h * 64 + hi * 8;
            bk0.u4[0] = *(const u32x4*)(pk);
            bk0.u4[1] = *(const u32x4*)(pk + 16);
            bk1.u4[0] = *(const u32x4*)(pk + 32);
            bk1.u4[1] = *(const u32x4*)(pk + 48);
            v8f acc = zero8();
            acc = wmma_bf16(aq[0], bk0, acc);
            acc = wmma_bf16(aq[1], bk1, acc);
#pragma unroll
            for (int r = 0; r < 8; r++)
                sc[(qsub * 16 + r + hi * 8) * 1024 + kb * 16 + laneRow] = acc[r];
        }
    }
    __syncthreads();

    // ---- Phase 2: softmax ---------------------------------------------
    {
        const float scale = 0.125f;   // 1/sqrt(64)
        for (int r = w * 4; r < w * 4 + 4; r++) {
            float mx = -3.4e38f;
            for (int c = lane; c < 1024; c += 32) {
                float v = sc[r * 1024 + c] * scale;
                if (mask[b * 1024 + c] == 0) v = -1e10f;
                sc[r * 1024 + c] = v;
                mx = fmaxf(mx, v);
            }
#pragma unroll
            for (int off = 16; off; off >>= 1) mx = fmaxf(mx, __shfl_xor(mx, off, 32));

            float sum = 0.f;
            for (int c = lane; c < 1024; c += 32) {
                float e = __expf(sc[r * 1024 + c] - mx);
                sc[r * 1024 + c] = e;
                sum += e;
            }
#pragma unroll
            for (int off = 16; off; off >>= 1) sum += __shfl_xor(sum, off, 32);
            const float inv = 1.0f / sum;

            float* arow = attn_out + ((size_t)((b * 16 + h) * 1024) + q0 + r) * 1024;
            for (int c = lane; c < 1024; c += 32) {
                float p = sc[r * 1024 + c] * inv;
                arow[c] = p;                 // the one mandatory S*S write
                pr[r * 1024 + c] = f2bf(p);  // kept in LDS for P@V
            }
        }
    }
    __syncthreads();

    // ---- Phase 3: X = P @ V  (2 accumulator chains) -------------------
    {
        const int qsub = w & 1;
        const int dsub = w >> 1;   // 0..3 -> 16-wide slice of head_dim
        v8f acc0 = zero8(), acc1 = zero8();
        const u16* vbase = Vt + ((size_t)(b * 16 + h) * 64 + dsub * 16 + laneRow) * 1024;
        const u16* pbase = pr + (qsub * 16 + laneRow) * 1024 + hi * 8;
        for (int kb = 0; kb < 32; kb += 2) {
            Frag ap0, bv0, ap1, bv1;
            const u16* pp0 = pbase + kb * 32;
            const u16* pp1 = pp0 + 32;
            ap0.u4[0] = *(const u32x4*)(pp0);
            ap0.u4[1] = *(const u32x4*)(pp0 + 16);
            ap1.u4[0] = *(const u32x4*)(pp1);
            ap1.u4[1] = *(const u32x4*)(pp1 + 16);
            const u16* pv0 = vbase + kb * 32 + hi * 8;
            const u16* pv1 = pv0 + 32;
            bv0.u4[0] = *(const u32x4*)(pv0);
            bv0.u4[1] = *(const u32x4*)(pv0 + 16);
            bv1.u4[0] = *(const u32x4*)(pv1);
            bv1.u4[1] = *(const u32x4*)(pv1 + 16);
            acc0 = wmma_bf16(ap0, bv0, acc0);
            acc1 = wmma_bf16(ap1, bv1, acc1);
        }
        v8f acc = acc0 + acc1;
#pragma unroll
        for (int r = 0; r < 8; r++) {
            const int row = q0 + qsub * 16 + r + hi * 8;
            const int col = h * 64 + dsub * 16 + laneRow;
            Xc[(size_t)(b * 1024 + row) * 1024 + col] = f2bf(acc[r]);
        }
    }
}

// ---------------------------------------------------------------------------
extern "C" void kernel_launch(void* const* d_in, const int* in_sizes, int n_in,
                              void* d_out, int out_size, void* d_ws, size_t ws_size,
                              hipStream_t stream) {
    const float* query = (const float*)d_in[0];
    const float* key   = (const float*)d_in[1];
    const float* value = (const float*)d_in[2];
    const int*   mask  = (const int*)d_in[3];
    const float* Wq    = (const float*)d_in[4];
    const float* bq    = (const float*)d_in[5];
    const float* Wk    = (const float*)d_in[6];
    const float* bk    = (const float*)d_in[7];
    const float* Wv    = (const float*)d_in[8];
    const float* bv    = (const float*)d_in[9];
    const float* Wo    = (const float*)d_in[10];
    const float* bo    = (const float*)d_in[11];

    float* out      = (float*)d_out;
    float* attn_out = out + (size_t)8 * 1024 * 1024;   // B*S*D

    // bf16 workspace layout (elements of u16); total 68M u16 = 136 MB
    const size_t NBIG = (size_t)8192 * 1024;   // B*S*D
    const size_t NW   = (size_t)1024 * 1024;
    u16* qb  = (u16*)d_ws;
    u16* kb  = qb  + NBIG;
    u16* vb  = kb  + NBIG;
    u16* Wqb = vb  + NBIG;
    u16* Wkb = Wqb + NW;
    u16* Wvb = Wkb + NW;
    u16* Wob = Wvb + NW;
    u16* Qp  = Wob + NW;
    u16* Kp  = Qp  + NBIG;
    u16* Vp  = Kp  + NBIG;
    u16* Vt  = Vp  + NBIG;
    u16* Xc  = Vt  + NBIG;

    // 1) fp32 -> bf16 (4 elements per thread)
    cvt_f32_bf16<<<NBIG / 1024, 256, 0, stream>>>(query, qb, (int)NBIG);
    cvt_f32_bf16<<<NBIG / 1024, 256, 0, stream>>>(key,   kb, (int)NBIG);
    cvt_f32_bf16<<<NBIG / 1024, 256, 0, stream>>>(value, vb, (int)NBIG);
    cvt_f32_bf16<<<NW   / 1024, 256, 0, stream>>>(Wq, Wqb, (int)NW);
    cvt_f32_bf16<<<NW   / 1024, 256, 0, stream>>>(Wk, Wkb, (int)NW);
    cvt_f32_bf16<<<NW   / 1024, 256, 0, stream>>>(Wv, Wvb, (int)NW);
    cvt_f32_bf16<<<NW   / 1024, 256, 0, stream>>>(Wo, Wob, (int)NW);

    // 2) Q/K/V projections (bf16 out), TDM-staged W tiles
    dim3 gg(8192 / 256, 1024 / 64);
    gemm_nt<<<gg, 256, 0, stream>>>(qb, Wqb, bq, Qp, nullptr, 8192, 1024, 1024);
    gemm_nt<<<gg, 256, 0, stream>>>(kb, Wkb, bk, Kp, nullptr, 8192, 1024, 1024);
    gemm_nt<<<gg, 256, 0, stream>>>(vb, Wvb, bv, Vp, nullptr, 8192, 1024, 1024);

    // 3) V transpose for K-contiguous PV fragments
    transpose_v<<<(int)(NBIG / 256), 256, 0, stream>>>(Vp, Vt);

    // 4) attention core (192 KB dynamic LDS per block)
    size_t smem = 32 * 1024 * sizeof(float) + 32 * 1024 * sizeof(u16);
    attn_kernel<<<4096, 256, smem, stream>>>(Qp, Kp, Vt, mask, attn_out, Xc);

    // 5) output projection (f32 out)
    gemm_nt<<<gg, 256, 0, stream>>>(Xc, Wob, bo, nullptr, out, 8192, 1024, 1024);
}
